// MSINCell_2645699854540
// MI455X (gfx1250) — compile-verified, matrix operand order checked by reading, not run on hardware
//
#include <hip/hip_runtime.h>
#include <cstdint>

// ---------------------------------------------------------------------------
// MSIN cell forward for MI455X (gfx1250, wave32, WMMA + TDM async staging).
// ---------------------------------------------------------------------------

#define NB    8192   // batch
#define NU    256    // units
#define NM    50     // messages
#define NDIN  512    // input dim

typedef __attribute__((ext_vector_type(16))) __bf16 v16bf;
typedef __attribute__((ext_vector_type(8)))  float  v8f;
typedef __attribute__((ext_vector_type(4)))  uint32_t u32x4;
typedef __attribute__((ext_vector_type(8)))  uint32_t u32x8;

union FragBF { v16bf v; uint4 q[2]; };

__device__ __forceinline__ __bf16 f2bf(float f) {
  unsigned u = __builtin_bit_cast(unsigned, f);
  unsigned r = (u + 0x7FFFu + ((u >> 16) & 1u)) >> 16;     // RNE
  unsigned short s = (unsigned short)r;
  return __builtin_bit_cast(__bf16, s);
}
__device__ __forceinline__ float bf2f(__bf16 b) {
  unsigned short s = __builtin_bit_cast(unsigned short, b);
  unsigned u = ((unsigned)s) << 16;
  return __builtin_bit_cast(float, u);
}
__device__ __forceinline__ float sigm(float x) { return 1.0f / (1.0f + __expf(-x)); }

// LDS byte offset of a __shared__ pointer: ISA 10.2 -- in the LDS aperture the
// flat address's low 32 bits ARE the workgroup-relative LDS offset.
__device__ __forceinline__ uint32_t lds_off(const void* p) {
  return (uint32_t)(uintptr_t)p;
}

// ---------------------------------------------------------------------------
// TDM: async 2D tile load (bf16), 64 elems wide x 16 rows, with LDS padding of
// 4 DWORDs after every 32 DWORDs (128B row) -> 144B padded LDS row stride.
// D# layout per CDNA5 ISA (group0: 4 SGPRs, group1: 8 SGPRs).
// ---------------------------------------------------------------------------
__device__ __forceinline__ void tdm_load_tile16(uint32_t lds_addr,
                                                const __bf16* gptr,
                                                uint32_t stride_elems) {
  uint64_t ga = (uint64_t)(uintptr_t)gptr;
  u32x4 g0;
  g0[0] = 1u;                                   // count=1, no gather, user mode
  g0[1] = lds_addr;                             // lds_addr [63:32]
  g0[2] = (uint32_t)ga;                         // global_addr[31:0]
  g0[3] = (uint32_t)((ga >> 32) & 0x01FFFFFFu)  // global_addr[56:32]
          | (2u << 30);                         // type = 2 ("image")
  u32x8 g1;
  g1[0] = (1u << 16)       // data_size = 1 (2 bytes)
        | (1u << 20)       // pad_enable
        | (4u << 22)       // pad_interval: code 4 -> every 32 DWORDs (128B)
        | (3u << 25);      // pad_amount:   code 3 -> 4 DWORDs (16B)
  g1[1] = (64u & 0xFFFFu) << 16;                // tensor_dim0 = 64 (low 16 in [31:16])
  g1[2] = (64u >> 16) | (16u << 16);            // tensor_dim0 hi | tensor_dim1 = 16 lo
  g1[3] = (16u >> 16) | (64u << 16);            // tensor_dim1 hi | tile_dim0 = 64
  g1[4] = 16u;                                  // tile_dim1 = 16, tile_dim2 = 0
  g1[5] = stride_elems;                         // tensor_dim0_stride[31:0]
  g1[6] = 0u;                                   // stride hi | dim1_stride lo
  g1[7] = 0u;
  asm volatile("tensor_load_to_lds %0, %1" :: "s"(g0), "s"(g1) : "memory");
}

// ---------------------------------------------------------------------------
// Prep kernels (tiny, run once per launch)
// ---------------------------------------------------------------------------

// dst[(n+nOff)*dstStride + kOff + k] = bf16(src[k*Nw + n])   (transpose+convert)
__global__ void tconv_kernel(__bf16* __restrict__ dst, int dstStride, int kOff, int nOff,
                             const float* __restrict__ src, int Kw, int Nw) {
  int idx = blockIdx.x * 256 + threadIdx.x;
  if (idx >= Kw * Nw) return;
  int k = idx / Nw, n = idx % Nw;
  dst[(size_t)(n + nOff) * dstStride + kOff + k] = f2bf(src[idx]);
}

__global__ void zero_bf16_kernel(__bf16* __restrict__ dst, int count) {
  int i = blockIdx.x * 256 + threadIdx.x;
  if (i < count) dst[i] = f2bf(0.0f);
}

__global__ void concat3_kernel(float* __restrict__ dst, const float* __restrict__ a,
                               const float* __restrict__ b, const float* __restrict__ c, int n) {
  int i = blockIdx.x * 256 + threadIdx.x;
  if (i >= 3 * n) return;
  dst[i] = (i < n) ? a[i] : (i < 2 * n ? b[i - n] : c[i - 2 * n]);
}

__global__ void cvt_bf16_kernel(__bf16* __restrict__ dst, const float* __restrict__ src, int n) {
  int i = blockIdx.x * 256 + threadIdx.x;
  if (i < n) dst[i] = f2bf(src[i]);
}

// A1 = [text | h]  bf16 [NB, 512]
__global__ void buildA1_kernel(__bf16* __restrict__ dst, const float* __restrict__ text,
                               const float* __restrict__ h) {
  int idx = blockIdx.x * 256 + threadIdx.x;
  if (idx >= NB * 512) return;
  int r = idx >> 9, c = idx & 511;
  float v = (c < 256) ? text[r * 256 + c] : h[r * 256 + (c - 256)];
  dst[idx] = f2bf(v);
}

// A2 = [X | h | V]  bf16 [NB, 1024]
__global__ void buildA2_kernel(__bf16* __restrict__ dst, const float* __restrict__ X,
                               const float* __restrict__ h, const float* __restrict__ V) {
  int idx = blockIdx.x * 256 + threadIdx.x;
  if (idx >= NB * 1024) return;
  int r = idx >> 10, c = idx & 1023;
  float v;
  if (c < 512)      v = X[(size_t)r * 512 + c];
  else if (c < 768) v = h[(size_t)r * 256 + (c - 512)];
  else              v = V[(size_t)r * 256 + (c - 768)];
  dst[idx] = f2bf(v);
}

// ---------------------------------------------------------------------------
// Generic bf16 WMMA GEMM:  C[M,N] = A[M,K](bf16, lda) * BT[N,K]^T + bias
// Tile 128x128x64, 256 threads = 8 waves, each wave -> 2x4 tiles of 16x16.
// Double-buffered LDS, staged by the Tensor Data Mover (each wave DMAs its own
// 16-row slice of the A and B tiles; TENSORcnt-based overlap with WMMA).
// ---------------------------------------------------------------------------
#define TBM 128
#define TBN 128
#define TBK 64
#define LDS_STRIDE 144   // 64 bf16 = 128B + 16B TDM pad -> conflict-free b128 reads

__global__ __launch_bounds__(256) void gemm_bf16_kernel(
    float* __restrict__ C, const __bf16* __restrict__ A, int lda,
    const __bf16* __restrict__ BT, const float* __restrict__ bias,
    int Mr, int N, int K) {
  __shared__ __align__(16) char sA[2][TBM * LDS_STRIDE];
  __shared__ __align__(16) char sB[2][TBN * LDS_STRIDE];
  (void)Mr;

  int ntiles = N / TBN;
  int bm = blockIdx.x / ntiles;
  int bn = blockIdx.x % ntiles;
  int t = threadIdx.x;
  int lane = t & 31, w = t >> 5;
  int wm = w & 3;    // 4 bands of 32 rows (2 m-tiles)
  int wn = w >> 2;   // 2 bands of 64 cols (4 n-tiles)
  int l16 = lane & 15;
  int lhalf = (lane < 16) ? 0 : 16;

  const int rowBase = bm * TBM;
  const int colBase = bn * TBN;
  const int nK = K / TBK;

  // Each wave stages rows [w*16, w*16+16) of both tiles via TDM.
  const __bf16* gA = A + (size_t)(rowBase + w * 16) * lda;
  const __bf16* gB = BT + (size_t)(colBase + w * 16) * K;
  uint32_t ldsA[2] = { lds_off(&sA[0][0]) + (uint32_t)(w * 16 * LDS_STRIDE),
                       lds_off(&sA[1][0]) + (uint32_t)(w * 16 * LDS_STRIDE) };
  uint32_t ldsB[2] = { lds_off(&sB[0][0]) + (uint32_t)(w * 16 * LDS_STRIDE),
                       lds_off(&sB[1][0]) + (uint32_t)(w * 16 * LDS_STRIDE) };

  v8f c[2][4];
#pragma unroll
  for (int mt = 0; mt < 2; ++mt)
#pragma unroll
    for (int nt = 0; nt < 4; ++nt)
#pragma unroll
      for (int i = 0; i < 8; ++i) c[mt][nt][i] = 0.0f;

  // prologue: DMA first tile pair into buffer 0
  tdm_load_tile16(ldsA[0], gA, (uint32_t)lda);
  tdm_load_tile16(ldsB[0], gB, (uint32_t)K);

  for (int kb = 0; kb < nK; ++kb) {
    int cur = kb & 1;
    if (kb + 1 < nK) {   // prefetch next tile pair into the other buffer
      tdm_load_tile16(ldsA[cur ^ 1], gA + (kb + 1) * TBK, (uint32_t)lda);
      tdm_load_tile16(ldsB[cur ^ 1], gB + (kb + 1) * TBK, (uint32_t)K);
      __builtin_amdgcn_s_wait_tensorcnt(2);   // current pair complete (in-order)
    } else {
      __builtin_amdgcn_s_wait_tensorcnt(0);
    }
    __syncthreads();     // all waves' slices of `cur` visible to everyone

    const char* bufA = sA[cur];
    const char* bufB = sB[cur];
#pragma unroll
    for (int kk = 0; kk < 2; ++kk) {    // two K=32 steps per TBK
      int kByte = kk * 64;
      FragBF bfr[4];
#pragma unroll
      for (int nt = 0; nt < 4; ++nt) {
        const char* p = bufB + (wn * 64 + nt * 16 + l16) * LDS_STRIDE + kByte;
        bfr[nt].q[0] = *(const uint4*)(p + lhalf);
        bfr[nt].q[1] = *(const uint4*)(p + lhalf + 32);
      }
#pragma unroll
      for (int mt = 0; mt < 2; ++mt) {
        FragBF afr;
        const char* p = bufA + (wm * 32 + mt * 16 + l16) * LDS_STRIDE + kByte;
        afr.q[0] = *(const uint4*)(p + lhalf);
        afr.q[1] = *(const uint4*)(p + lhalf + 32);
#pragma unroll
        for (int nt = 0; nt < 4; ++nt)
          c[mt][nt] = __builtin_amdgcn_wmma_f32_16x16x32_bf16(
              false, afr.v, false, bfr[nt].v, (short)0, c[mt][nt], false, false);
      }
    }
    __syncthreads();     // everyone done reading `cur` before it is DMA-reused
  }

  // epilogue: C layout per ISA -- VGPR i: lanes 0-15 -> M=i, lanes 16-31 -> M=i+8
#pragma unroll
  for (int mt = 0; mt < 2; ++mt)
#pragma unroll
    for (int nt = 0; nt < 4; ++nt) {
      int col = colBase + wn * 64 + nt * 16 + l16;
      float bv = bias ? bias[col] : 0.0f;
      int m0 = rowBase + wm * 32 + mt * 16 + ((lane < 16) ? 0 : 8);
#pragma unroll
      for (int i = 0; i < 8; ++i)
        C[(size_t)(m0 + i) * N + col] = c[mt][nt][i] + bv;
    }
}

// ---------------------------------------------------------------------------
// Fused attention: one workgroup per batch element.
//   a = tanh(S[b] @ W_sa + hb[b]);  scores = a . v_a;  P = softmax;
//   text[b] = P @ S[b].   S[b] staged ONCE to LDS (bf16, padded to 64 rows).
// (manual staging kept: TDM cannot do the fp32 -> bf16 conversion)
// ---------------------------------------------------------------------------
#define S_STRIDE 528   // 256 bf16 = 512B + 16B pad

__global__ __launch_bounds__(256) void attn_kernel(
    const float* __restrict__ S, const __bf16* __restrict__ WsaT,
    const float* __restrict__ hb, const float* __restrict__ v_a,
    float* __restrict__ Pout, float* __restrict__ text) {
  __shared__ __align__(16) char sS[64 * S_STRIDE];
  __shared__ float score[64];
  __shared__ float Pl[64];

  int b = blockIdx.x;
  int t = threadIdx.x;
  if (t < 64) score[t] = 0.0f;

  const float* Sb = S + (size_t)b * NM * NU;
  for (int r = 0; r < 64; ++r) {
    float v = (r < NM) ? Sb[r * NU + t] : 0.0f;
    *(__bf16*)(sS + r * S_STRIDE + t * 2) = f2bf(v);
  }
  __syncthreads();

  int lane = t & 31, w = t >> 5;
  int wm = w & 1;    // 2 bands of 32 rows over 64
  int wn = w >> 1;   // 4 bands of 64 cols over 256
  int l16 = lane & 15;
  int lhalf = (lane < 16) ? 0 : 16;

  v8f c[2][4];
#pragma unroll
  for (int mt = 0; mt < 2; ++mt)
#pragma unroll
    for (int nt = 0; nt < 4; ++nt)
#pragma unroll
      for (int i = 0; i < 8; ++i) c[mt][nt][i] = 0.0f;

#pragma unroll
  for (int kc = 0; kc < 8; ++kc) {     // K = 256, 32 per WMMA
    int kByte = kc * 64;
    FragBF bfr[4];
#pragma unroll
    for (int nt = 0; nt < 4; ++nt) {   // B frags straight from L2-resident WsaT[N,K]
      const char* p = (const char*)(WsaT + (size_t)(wn * 64 + nt * 16 + l16) * NU + kc * 32);
      bfr[nt].q[0] = *(const uint4*)(p + lhalf);
      bfr[nt].q[1] = *(const uint4*)(p + lhalf + 32);
    }
#pragma unroll
    for (int mt = 0; mt < 2; ++mt) {
      FragBF afr;
      const char* p = sS + (wm * 32 + mt * 16 + l16) * S_STRIDE + kByte;
      afr.q[0] = *(const uint4*)(p + lhalf);
      afr.q[1] = *(const uint4*)(p + lhalf + 32);
#pragma unroll
      for (int nt = 0; nt < 4; ++nt)
        c[mt][nt] = __builtin_amdgcn_wmma_f32_16x16x32_bf16(
            false, afr.v, false, bfr[nt].v, (short)0, c[mt][nt], false, false);
    }
  }

  // scores[m] += tanh(c + hb) * v_a  (LDS float atomics)
  const float* hbb = hb + (size_t)b * NU;
#pragma unroll
  for (int nt = 0; nt < 4; ++nt) {
    int col = wn * 64 + nt * 16 + l16;
    float hbv = hbb[col];
    float vav = v_a[col];
#pragma unroll
    for (int mt = 0; mt < 2; ++mt) {
      int m0 = wm * 32 + mt * 16 + ((lane < 16) ? 0 : 8);
#pragma unroll
      for (int i = 0; i < 8; ++i) {
        float a = tanhf(c[mt][nt][i] + hbv);
        atomicAdd(&score[m0 + i], a * vav);
      }
    }
  }
  __syncthreads();

  if (t == 0) {   // softmax over 50 messages (trivial)
    float mx = -1e30f;
    for (int m = 0; m < NM; ++m) mx = fmaxf(mx, score[m]);
    float s = 0.0f;
    for (int m = 0; m < NM; ++m) { float e = __expf(score[m] - mx); Pl[m] = e; s += e; }
    float inv = 1.0f / s;
    for (int m = 0; m < NM; ++m) Pl[m] *= inv;
  }
  __syncthreads();

  if (t < NM) Pout[(size_t)b * NM + t] = Pl[t];

  float acc = 0.0f;   // text[b,u] = sum_m P[m] * S[b,m,u] from LDS
  for (int m = 0; m < NM; ++m)
    acc += Pl[m] * bf2f(*(const __bf16*)(sS + m * S_STRIDE + t * 2));
  text[(size_t)b * NU + t] = acc;
}

// ---------------------------------------------------------------------------
// Elementwise epilogues
// ---------------------------------------------------------------------------
__global__ void ewV_kernel(float* __restrict__ Vout, const float* __restrict__ G,
                           const float* __restrict__ v) {
  int idx = blockIdx.x * 256 + threadIdx.x;
  if (idx >= NB * NU) return;
  int r = idx >> 8, u = idx & 255;
  const float* g = G + (size_t)r * 768;
  float F = sigm(g[u]);
  float O = sigm(g[u + 256]);
  float tn = tanhf(g[u + 512]);
  Vout[idx] = F * v[idx] + O * tn;
}

__global__ void ewH_kernel(float* __restrict__ Hout, const float* __restrict__ G) {
  int idx = blockIdx.x * 256 + threadIdx.x;
  if (idx >= NB * NU) return;
  int r = idx >> 8, u = idx & 255;
  const float* g = G + (size_t)r * 768;
  float k  = sigm(g[u]);
  float hx = tanhf(g[u + 256]);
  float hv = tanhf(g[u + 512]);
  Hout[idx] = (1.0f - k) * hv + k * hx;
}

// ---------------------------------------------------------------------------
// Launcher
// ---------------------------------------------------------------------------
extern "C" void kernel_launch(void* const* d_in, const int* in_sizes, int n_in,
                              void* d_out, int out_size, void* d_ws, size_t ws_size,
                              hipStream_t stream) {
  const float* X    = (const float*)d_in[0];
  const float* S    = (const float*)d_in[1];
  const float* h    = (const float*)d_in[2];
  const float* v    = (const float*)d_in[3];
  const float* W_sa = (const float*)d_in[4];
  const float* W_ha = (const float*)d_in[5];
  const float* b_a  = (const float*)d_in[6];
  const float* v_a  = (const float*)d_in[7];
  const float* W_f  = (const float*)d_in[8];
  const float* W_hf = (const float*)d_in[9];
  const float* b_f  = (const float*)d_in[10];
  const float* W_o  = (const float*)d_in[11];
  const float* W_ho = (const float*)d_in[12];
  const float* b_o  = (const float*)d_in[13];
  const float* W_t  = (const float*)d_in[14];
  const float* W_ht = (const float*)d_in[15];
  const float* b_t  = (const float*)d_in[16];
  const float* W_k  = (const float*)d_in[17];
  const float* W_hk = (const float*)d_in[18];
  const float* W_vk = (const float*)d_in[19];
  const float* b_k  = (const float*)d_in[20];
  const float* W_s  = (const float*)d_in[21];
  const float* W_hs = (const float*)d_in[22];
  const float* b_s  = (const float*)d_in[23];
  const float* W_v  = (const float*)d_in[24];
  const float* W_hv = (const float*)d_in[25];
  const float* b_v  = (const float*)d_in[26];

  float* outH = (float*)d_out;                 // [8192,256]
  float* outP = outH + (size_t)NB * NU;        // [8192,50]
  float* outV = outP + (size_t)NB * NM;        // [8192,256]

  char* wsb = (char*)d_ws;
  size_t off = 0;
  auto take = [&](size_t bytes) -> char* {
    char* p = wsb + off;
    off = (off + bytes + 255) & ~(size_t)255;
    return p;
  };
  __bf16* WsaT  = (__bf16*)take((size_t)256 * 256 * 2);
  __bf16* WhaT  = (__bf16*)take((size_t)256 * 256 * 2);
  __bf16* WgT   = (__bf16*)take((size_t)768 * 512 * 2);    // [text|h] gate weights
  __bf16* WkhvT = (__bf16*)take((size_t)768 * 1024 * 2);   // [X|h|V] block weights
  float*  biasG = (float*)take(768 * 4);
  float*  biasK = (float*)take(768 * 4);
  __bf16* hbf   = (__bf16*)take((size_t)NB * 256 * 2);
  float*  hb    = (float*)take((size_t)NB * 256 * 4);
  float*  text  = (float*)take((size_t)NB * 256 * 4);
  __bf16* A1    = (__bf16*)take((size_t)NB * 512 * 2);
  float*  Gbuf  = (float*)take((size_t)NB * 768 * 4);
  __bf16* A2    = (__bf16*)take((size_t)NB * 1024 * 2);
  float*  KHV   = (float*)take((size_t)NB * 768 * 4);
  (void)ws_size; (void)in_sizes; (void)n_in; (void)out_size;

  auto tcGrid = [](int Kw, int Nw) { return dim3((Kw * Nw + 255) / 256); };

  // ---- weight prep (bf16, transposed to [N][K]) ----
  tconv_kernel<<<tcGrid(256, 256), 256, 0, stream>>>(WsaT, 256, 0, 0, W_sa, 256, 256);
  tconv_kernel<<<tcGrid(256, 256), 256, 0, stream>>>(WhaT, 256, 0, 0, W_ha, 256, 256);
  tconv_kernel<<<tcGrid(256, 256), 256, 0, stream>>>(WgT, 512, 0,   0,   W_f,  256, 256);
  tconv_kernel<<<tcGrid(256, 256), 256, 0, stream>>>(WgT, 512, 256, 0,   W_hf, 256, 256);
  tconv_kernel<<<tcGrid(256, 256), 256, 0, stream>>>(WgT, 512, 0,   256, W_o,  256, 256);
  tconv_kernel<<<tcGrid(256, 256), 256, 0, stream>>>(WgT, 512, 256, 256, W_ho, 256, 256);
  tconv_kernel<<<tcGrid(256, 256), 256, 0, stream>>>(WgT, 512, 0,   512, W_t,  256, 256);
  tconv_kernel<<<tcGrid(256, 256), 256, 0, stream>>>(WgT, 512, 256, 512, W_ht, 256, 256);
  zero_bf16_kernel<<<(768 * 1024 + 255) / 256, 256, 0, stream>>>(WkhvT, 768 * 1024);
  tconv_kernel<<<tcGrid(512, 256), 256, 0, stream>>>(WkhvT, 1024, 0,   0,   W_k,  512, 256);
  tconv_kernel<<<tcGrid(256, 256), 256, 0, stream>>>(WkhvT, 1024, 512, 0,   W_hk, 256, 256);
  tconv_kernel<<<tcGrid(256, 256), 256, 0, stream>>>(WkhvT, 1024, 768, 0,   W_vk, 256, 256);
  tconv_kernel<<<tcGrid(512, 256), 256, 0, stream>>>(WkhvT, 1024, 0,   256, W_s,  512, 256);
  tconv_kernel<<<tcGrid(256, 256), 256, 0, stream>>>(WkhvT, 1024, 512, 256, W_hs, 256, 256);
  tconv_kernel<<<tcGrid(256, 256), 256, 0, stream>>>(WkhvT, 1024, 512, 512, W_hv, 256, 256);
  tconv_kernel<<<tcGrid(256, 256), 256, 0, stream>>>(WkhvT, 1024, 768, 512, W_v,  256, 256);
  concat3_kernel<<<3, 256, 0, stream>>>(biasG, b_f, b_o, b_t, 256);
  concat3_kernel<<<3, 256, 0, stream>>>(biasK, b_k, b_s, b_v, 256);
  cvt_bf16_kernel<<<(NB * 256 + 255) / 256, 256, 0, stream>>>(hbf, h, NB * 256);

  // ---- hb = h @ W_ha + b_a ----
  gemm_bf16_kernel<<<(NB / TBM) * (256 / TBN), 256, 0, stream>>>(
      hb, hbf, 256, WhaT, b_a, NB, 256, 256);

  // ---- fused attention: scores/softmax/P/text (S read once) ----
  attn_kernel<<<NB, 256, 0, stream>>>(S, WsaT, hb, v_a, outP, text);

  // ---- gates: [text|h] @ [Wf;Whf | Wo;Who | Wt;Wht] + bias ----
  buildA1_kernel<<<(NB * 512 + 255) / 256, 256, 0, stream>>>(A1, text, h);
  gemm_bf16_kernel<<<(NB / TBM) * (768 / TBN), 256, 0, stream>>>(
      Gbuf, A1, 512, WgT, biasG, NB, 768, 512);
  ewV_kernel<<<(NB * NU + 255) / 256, 256, 0, stream>>>(outV, Gbuf, v);

  // ---- k/hx/hv: [X|h|V] @ block weights + bias ----
  buildA2_kernel<<<(NB * 1024 + 255) / 256, 256, 0, stream>>>(A2, X, h, outV);
  gemm_bf16_kernel<<<(NB / TBM) * (768 / TBN), 256, 0, stream>>>(
      KHV, A2, 1024, WkhvT, biasK, NB, 768, 1024);
  ewH_kernel<<<(NB * NU + 255) / 256, 256, 0, stream>>>(outH, KHV);
}
